// MultiScaleRetention_24438363914602
// MI455X (gfx1250) — matrix operand order repaired and checked by reading, not compiled, not run on hardware
//
#include <hip/hip_runtime.h>

#define T_LEN 2048
#define E_DIM 2048
#define H_NUM 16
#define KD_DIM 128
#define V_DIM 3072
#define HD_DIM 192
#define EPS_F 1e-6f

typedef __attribute__((ext_vector_type(16))) __bf16 v16bf;
typedef __attribute__((ext_vector_type(8)))  __bf16 v8bf;
typedef __attribute__((ext_vector_type(8)))  float  v8f;
typedef __attribute__((ext_vector_type(4)))  unsigned int u32x4;
typedef __attribute__((ext_vector_type(8)))  int i32x8;
typedef __attribute__((ext_vector_type(4)))  int i32x4;

// ---------------- fragment loaders (wave32, 16x16x32 bf16 WMMA) ----------------
static __device__ inline v16bf make_frag(const __bf16* p0, const __bf16* p1) {
  union { v16bf v; v8bf h[2]; } u;
  u.h[0] = *(const v8bf*)p0;
  u.h[1] = *(const v8bf*)p1;
  return u.v;
}
// A fragment (16x32 MxK): lane<16 row M=lane&15, K {0..7,16..23}; lane>=16 K {8..15,24..31}
static __device__ inline v16bf load_a_frag(const __bf16* rowp, int lane) {
  int kb = (lane >> 4) << 3;                   // 0 or 8
  return make_frag(rowp + kb, rowp + 16 + kb);
}
// B fragment (32x16 KxN) from n-major (transposed) storage: lane holds N=lane&15,
// K {0..7,16..23} for lane<16; {8..15,24..31} for lane>=16 (matches A)
static __device__ inline v16bf load_b_frag(const __bf16* rowp, int lane) {
  const __bf16* p = rowp + ((lane >> 4) << 3);  // +0 or +8
  return make_frag(p, p + 16);
}
static __device__ inline v8f wmma_bf16(v16bf a, v16bf b, v8f c) {
  return __builtin_amdgcn_wmma_f32_16x16x32_bf16(false, a, false, b, (short)0, c, false, false);
}

// ---------------- Tensor Data Mover: 2D bf16 tile (tile1 lines x tile0 elems) -> LDS ----------------
// D# per CDNA5 ISA: group0 = {count, lds_addr, global_addr, type=2},
// group1 = {data_size=2B, tensor_dim0/1, tile_dim0/1, tensor_dim0_stride}.
static __device__ inline void tdm_load_2d(unsigned lds_addr, const void* gptr,
                                          unsigned tile0, unsigned tile1,
                                          unsigned stride0_elems) {
  unsigned long long ga = (unsigned long long)gptr;
  u32x4 g0 = { 1u,                                   // count = 1 valid descriptor
               lds_addr,                             // [63:32] LDS byte address
               (unsigned)ga,                         // [95:64] global addr lo
               (unsigned)((ga >> 32) & 0x1ffffffu) | (2u << 30) }; // addr hi | type=2
  unsigned dim0 = stride0_elems;                     // tensor extent dim0 (elements)
  unsigned dim1 = 0x7fffffffu;                       // effectively unbounded dim1
  i32x8 g1 = { 0x10000,                              // data_size = 1 (2-byte elements)
               (int)((dim0 & 0xffffu) << 16),        // tensor_dim0[15:0]
               (int)((dim0 >> 16) | ((dim1 & 0xffffu) << 16)),
               (int)((dim1 >> 16) | (tile0 << 16)),  // tile_dim0
               (int)(tile1 & 0xffffu),               // tile_dim1
               (int)stride0_elems,                   // tensor_dim0_stride[31:0]
               0, 0 };
  i32x4 z4 = {0, 0, 0, 0};
#if __clang_major__ >= 23
  i32x8 z8 = {0, 0, 0, 0, 0, 0, 0, 0};
  __builtin_amdgcn_tensor_load_to_lds(g0, g1, z4, z4, z8, 0);
#else
  __builtin_amdgcn_tensor_load_to_lds(g0, g1, z4, z4, 0);
#endif
}

// ---------------- elementwise f32 -> bf16 ----------------
__global__ void f32_to_bf16_kernel(const float* __restrict__ src, __bf16* __restrict__ dst, size_t n) {
  size_t i = (size_t)blockIdx.x * blockDim.x + threadIdx.x;
  if (i < n) dst[i] = (__bf16)src[i];
}

// ---------------- tiled transpose + convert: W (rows x cols f32) -> Wt (cols x rows bf16) ----------------
__global__ __launch_bounds__(256) void transpose_f32_to_bf16(
    const float* __restrict__ src, __bf16* __restrict__ dst, int rows, int cols)
{
  __shared__ float tile[32][33];
  const int nbc = cols / 32;
  const int r0 = (blockIdx.x / nbc) * 32;
  const int c0 = (blockIdx.x % nbc) * 32;
  const int tx = threadIdx.x & 31, ty = threadIdx.x >> 5;   // 32x8
  #pragma unroll
  for (int i = 0; i < 32; i += 8)
    tile[ty + i][tx] = src[(size_t)(r0 + ty + i) * cols + c0 + tx];
  __syncthreads();
  #pragma unroll
  for (int i = 0; i < 32; i += 8)
    dst[(size_t)(c0 + ty + i) * rows + r0 + tx] = (__bf16)tile[tx][ty + i];
}

// ---------------- generic bf16 GEMM: C(f32 MxN) = A(MxK, row-major) * Bt(NxK, n-major) ----------------
// block tile 128x64, 4 waves; B tiles double-buffered in LDS via TDM (pipelined on TENSORcnt),
// A fragments direct from global.
template<int N, int K>
__global__ __launch_bounds__(128) void gemm_bf16_kernel(
    const __bf16* __restrict__ A, const __bf16* __restrict__ Btg, float* __restrict__ C)
{
  __shared__ __bf16 Bt[2][64][32];
  const int tid = threadIdx.x, lane = tid & 31, wave = tid >> 5;
  const int m0 = (blockIdx.x / (N / 64)) * 128;
  const int n0 = (blockIdx.x % (N / 64)) * 64;

  const v8f zero = {0.f,0.f,0.f,0.f,0.f,0.f,0.f,0.f};
  v8f acc[2][4];
  #pragma unroll
  for (int rt = 0; rt < 2; ++rt)
    #pragma unroll
    for (int t = 0; t < 4; ++t) acc[rt][t] = zero;

  const __bf16* arow0 = A + (size_t)(m0 + wave * 32 + (lane & 15)) * K;
  const __bf16* arow1 = arow0 + (size_t)16 * K;
  const __bf16* bbase = Btg + (size_t)n0 * K;
  const unsigned ldsB0 = (unsigned)(unsigned long long)(uintptr_t)&Bt[0][0][0];
  const unsigned ldsB1 = (unsigned)(unsigned long long)(uintptr_t)&Bt[1][0][0];

  constexpr int iters = K / 32;
  if (wave == 0)
    tdm_load_2d(ldsB0, bbase, 32u, 64u, (unsigned)K);     // prefetch tile 0

  for (int i = 0; i < iters; ++i) {
    const int k0 = i * 32;
    __syncthreads();                       // buffer (i+1)&1 free for overwrite
    if (wave == 0) {
      if (i + 1 < iters) {
        tdm_load_2d(((i + 1) & 1) ? ldsB1 : ldsB0, bbase + k0 + 32, 32u, 64u, (unsigned)K);
        __builtin_amdgcn_s_wait_tensorcnt((short)1);  // current tile landed; next in flight
      } else {
        __builtin_amdgcn_s_wait_tensorcnt((short)0);
      }
    }
    __syncthreads();
    const int buf = i & 1;
    v16bf a0 = load_a_frag(arow0 + k0, lane);
    v16bf a1 = load_a_frag(arow1 + k0, lane);
    #pragma unroll
    for (int t = 0; t < 4; ++t) {
      v16bf b = load_b_frag(&Bt[buf][t * 16 + (lane & 15)][0], lane);
      acc[0][t] = wmma_bf16(a0, b, acc[0][t]);
      acc[1][t] = wmma_bf16(a1, b, acc[1][t]);
    }
  }
  const int ncol = lane & 15;
  #pragma unroll
  for (int rt = 0; rt < 2; ++rt) {
    const int mrow0 = m0 + wave * 32 + rt * 16 + (lane >> 4) * 8;
    float* cp = C + (size_t)mrow0 * N + n0 + ncol;
    #pragma unroll
    for (int t = 0; t < 4; ++t)
      #pragma unroll
      for (int r = 0; r < 8; ++r)
        cp[(size_t)r * N + t * 16] = acc[rt][t][r];   // r*N folds to immediate (N constexpr)
  }
}

// ---------------- RoPE: q,k (T x E f32) -> qr,kr bf16 in [H][T][KD] ----------------
__global__ void rope_kernel(const float* __restrict__ q, const float* __restrict__ k,
                            const float* __restrict__ sn, const float* __restrict__ cs,
                            __bf16* __restrict__ qr, __bf16* __restrict__ kr)
{
  const float kscale = 0.088388347648318447f;   // 1/sqrt(KD)
  size_t idx = (size_t)blockIdx.x * blockDim.x + threadIdx.x;
  const int ppr = E_DIM / 2;
  int t  = (int)(idx / ppr);
  int e0 = 2 * (int)(idx % ppr);
  int h  = e0 / KD_DIM, d0 = e0 % KD_DIM;
  float s0 = sn[t * KD_DIM + d0], s1 = sn[t * KD_DIM + d0 + 1];
  float c0 = cs[t * KD_DIM + d0], c1 = cs[t * KD_DIM + d0 + 1];
  float q0 = q[(size_t)t * E_DIM + e0], q1 = q[(size_t)t * E_DIM + e0 + 1];
  float k0 = k[(size_t)t * E_DIM + e0], k1 = k[(size_t)t * E_DIM + e0 + 1];
  size_t o = ((size_t)h * T_LEN + t) * KD_DIM + d0;
  qr[o]     = (__bf16)(q0 * c0 - q1 * s0);
  qr[o + 1] = (__bf16)(q1 * c1 + q0 * s1);
  kr[o]     = (__bf16)(kscale * (k0 * c0 - k1 * s0));
  kr[o + 1] = (__bf16)(kscale * (k1 * c1 + k0 * s1));
}

// ---------------- V transpose: (T x V f32) -> bf16 [H][HD][T] ----------------
__global__ __launch_bounds__(256) void v_transpose_kernel(const float* __restrict__ v,
                                                          __bf16* __restrict__ vt)
{
  __shared__ float tile[32][33];
  const int tiles_per_head = (HD_DIM / 32) * (T_LEN / 32);
  const int h  = blockIdx.x / tiles_per_head;
  const int rr = blockIdx.x % tiles_per_head;
  const int hd0 = (rr % (HD_DIM / 32)) * 32;
  const int t0  = (rr / (HD_DIM / 32)) * 32;
  const int tx = threadIdx.x & 31, ty = threadIdx.x >> 5;
  #pragma unroll
  for (int i = 0; i < 32; i += 8)
    tile[ty + i][tx] = v[(size_t)(t0 + ty + i) * V_DIM + h * HD_DIM + hd0 + tx];
  __syncthreads();
  #pragma unroll
  for (int i = 0; i < 32; i += 8)
    vt[((size_t)h * HD_DIM + hd0 + ty + i) * T_LEN + t0 + tx] = (__bf16)tile[tx][ty + i];
}

// ---------------- retention: one block per (head, 64-row tile), flash-style single pass ----------------
// K/V blocks double-buffered in LDS via TDM, pipelined on TENSORcnt.
__global__ __launch_bounds__(128) void retention_kernel(
    const __bf16* __restrict__ qr, const __bf16* __restrict__ kr,
    const __bf16* __restrict__ vt, const float* __restrict__ dm,
    float* __restrict__ o)
{
  __shared__ __bf16 Kb[2][32][KD_DIM];   // 2 x 8 KB  key blocks (row-major)
  __shared__ __bf16 Vt[2][HD_DIM][32];   // 2 x 12 KB value blocks ([hd][s])
  __shared__ __bf16 Sb[4][16][32];       // 4 KB  per-wave S bounce buffer

  const int tid  = threadIdx.x, lane = tid & 31, wave = tid >> 5;
  const int h    = blockIdx.x >> 5;
  const int m0   = (blockIdx.x & 31) * 64;
  const int row  = m0 + wave * 16 + (lane & 15);

  const __bf16* qbase = qr + (size_t)h * T_LEN * KD_DIM;
  const __bf16* kbase = kr + (size_t)h * T_LEN * KD_DIM;
  const __bf16* vbase = vt + (size_t)h * HD_DIM * T_LEN;
  const float*  dmh   = dm + (size_t)h * T_LEN * T_LEN;
  const unsigned ldsK0 = (unsigned)(unsigned long long)(uintptr_t)&Kb[0][0][0];
  const unsigned ldsK1 = (unsigned)(unsigned long long)(uintptr_t)&Kb[1][0][0];
  const unsigned ldsV0 = (unsigned)(unsigned long long)(uintptr_t)&Vt[0][0][0];
  const unsigned ldsV1 = (unsigned)(unsigned long long)(uintptr_t)&Vt[1][0][0];

  v16bf qa[4];
  #pragma unroll
  for (int c = 0; c < 4; ++c)
    qa[c] = load_a_frag(qbase + (size_t)row * KD_DIM + 32 * c, lane);

  const v8f zero = {0.f,0.f,0.f,0.f,0.f,0.f,0.f,0.f};
  v8f oacc[12];
  #pragma unroll
  for (int t = 0; t < 12; ++t) oacc[t] = zero;
  float rowsum[8];
  #pragma unroll
  for (int r = 0; r < 8; ++r) rowsum[r] = 0.f;

  const int iters = (m0 + 64) / 32;      // causal: dm==0 above the diagonal
  if (wave == 0) {                       // prefetch block 0
    tdm_load_2d(ldsK0, kbase, (unsigned)KD_DIM, 32u, (unsigned)KD_DIM);
    tdm_load_2d(ldsV0, vbase, 32u, (unsigned)HD_DIM, (unsigned)T_LEN);
  }

  for (int i = 0; i < iters; ++i) {
    const int s0 = i * 32;
    __syncthreads();                     // buffer (i+1)&1 free for overwrite
    if (wave == 0) {
      if (i + 1 < iters) {
        const int nb = (i + 1) & 1;
        tdm_load_2d(nb ? ldsK1 : ldsK0, kbase + (size_t)(s0 + 32) * KD_DIM,
                    (unsigned)KD_DIM, 32u, (unsigned)KD_DIM);
        tdm_load_2d(nb ? ldsV1 : ldsV0, vbase + s0 + 32,
                    32u, (unsigned)HD_DIM, (unsigned)T_LEN);
        __builtin_amdgcn_s_wait_tensorcnt((short)2);  // current pair landed
      } else {
        __builtin_amdgcn_s_wait_tensorcnt((short)0);
      }
    }
    __syncthreads();
    const int buf = i & 1;

    // S = qr @ kr^T (two 16-wide subtiles), scale by decay mask, track |.| row sums
    #pragma unroll
    for (int sc = 0; sc < 2; ++sc) {
      v8f sf = zero;
      #pragma unroll
      for (int c = 0; c < 4; ++c) {
        v16bf b = load_b_frag(&Kb[buf][sc * 16 + (lane & 15)][32 * c], lane);
        sf = wmma_bf16(qa[c], b, sf);
      }
      const int mloc = (lane >> 4) * 8;
      const int scol = s0 + sc * 16 + (lane & 15);
      #pragma unroll
      for (int r = 0; r < 8; ++r) {
        float d   = dmh[(size_t)(m0 + wave * 16 + mloc + r) * T_LEN + scol];
        float val = sf[r] * d;
        rowsum[r] += fabsf(val);
        Sb[wave][mloc + r][sc * 16 + (lane & 15)] = (__bf16)val;
      }
    }
    asm volatile("s_wait_dscnt 0" ::: "memory");   // same-wave LDS store->load ordering

    // O += S @ V
    v16bf sa = load_a_frag(&Sb[wave][lane & 15][0], lane);
    #pragma unroll
    for (int t = 0; t < 12; ++t) {
      v16bf b = load_b_frag(&Vt[buf][t * 16 + (lane & 15)][0], lane);
      oacc[t] = wmma_bf16(sa, b, oacc[t]);
    }
  }

  // reduce |.| row sums across 16 lanes sharing a row (xor masks keep lane-half fixed)
  #pragma unroll
  for (int r = 0; r < 8; ++r) {
    float v = rowsum[r];
    #pragma unroll
    for (int mm = 1; mm < 16; mm <<= 1) v += __shfl_xor(v, mm, 32);
    rowsum[r] = fminf(fmaxf(v, 1.0f), 50000.0f);
  }

  const int mrow0 = m0 + wave * 16 + (lane >> 4) * 8;
  const int ncol  = lane & 15;
  #pragma unroll
  for (int t = 0; t < 12; ++t)
    #pragma unroll
    for (int r = 0; r < 8; ++r)
      o[(size_t)(mrow0 + r) * V_DIM + h * HD_DIM + t * 16 + ncol] = oacc[t][r] / rowsum[r];
}

// ---------------- groupnorm (rms over HD) + swish gate -> bf16 ----------------
__global__ __launch_bounds__(64) void gate_norm_kernel(
    const float* __restrict__ o, const float* __restrict__ g,
    const float* __restrict__ gn_w, __bf16* __restrict__ gated)
{
  __shared__ float red[64];
  const int t = blockIdx.x / H_NUM;
  const int h = blockIdx.x % H_NUM;
  const float* orow = o + (size_t)t * V_DIM + h * HD_DIM;
  const float* grow = g + (size_t)t * V_DIM + h * HD_DIM;

  float vals[3];
  float ss = 0.f;
  #pragma unroll
  for (int i = 0; i < 3; ++i) {
    float x = orow[threadIdx.x + 64 * i];
    vals[i] = x;
    ss += x * x;
  }
  red[threadIdx.x] = ss;
  __syncthreads();
  for (int off = 32; off >= 1; off >>= 1) {
    if (threadIdx.x < off) red[threadIdx.x] += red[threadIdx.x + off];
    __syncthreads();
  }
  const float inv = rsqrtf(red[0] / (float)HD_DIM + EPS_F);
  #pragma unroll
  for (int i = 0; i < 3; ++i) {
    int c = threadIdx.x + 64 * i;
    float gg = grow[c];
    float sw = gg / (1.f + expf(-gg));
    gated[(size_t)t * V_DIM + h * HD_DIM + c] = (__bf16)(sw * vals[i] * inv * gn_w[c]);
  }
}

// ---------------- host side ----------------
extern "C" void kernel_launch(void* const* d_in, const int* in_sizes, int n_in,
                              void* d_out, int out_size, void* d_ws, size_t ws_size,
                              hipStream_t stream) {
  (void)in_sizes; (void)n_in; (void)out_size; (void)ws_size;
  const float* hs   = (const float*)d_in[0];
  const float* Wq   = (const float*)d_in[1];
  const float* Wk   = (const float*)d_in[2];
  const float* Wv   = (const float*)d_in[3];
  const float* Wg   = (const float*)d_in[4];
  const float* Wout = (const float*)d_in[5];
  const float* gnw  = (const float*)d_in[6];
  const float* sn   = (const float*)d_in[7];
  const float* cs   = (const float*)d_in[8];
  const float* dm   = (const float*)d_in[9];
  float* out = (float*)d_out;

  size_t off = 0;
  auto alloc = [&](size_t bytes) -> char* {
    char* p = (char*)d_ws + off;
    off += (bytes + 255) & ~(size_t)255;
    return p;
  };
  const size_t TE = (size_t)T_LEN * E_DIM, TV = (size_t)T_LEN * V_DIM;
  const size_t EE = (size_t)E_DIM * E_DIM, EV = (size_t)E_DIM * V_DIM, VE = (size_t)V_DIM * E_DIM;

  __bf16* hs_bf  = (__bf16*)alloc(TE * 2);
  __bf16* wqt    = (__bf16*)alloc(EE * 2);   // [N=E][K=E]
  __bf16* wkt    = (__bf16*)alloc(EE * 2);
  __bf16* wvt    = (__bf16*)alloc(EV * 2);   // [N=V][K=E]
  __bf16* wgt    = (__bf16*)alloc(EV * 2);
  __bf16* woutt  = (__bf16*)alloc(VE * 2);   // [N=E][K=V]
  float*  q_f    = (float*)alloc(TE * 4);
  float*  k_f    = (float*)alloc(TE * 4);
  float*  v_f    = (float*)alloc(TV * 4);
  float*  g_f    = (float*)alloc(TV * 4);
  __bf16* qr_bf  = (__bf16*)alloc(TE * 2);
  __bf16* kr_bf  = (__bf16*)alloc(TE * 2);
  __bf16* vt_bf  = (__bf16*)alloc(TV * 2);   // [H][HD][T]
  float*  o_f    = (float*)alloc(TV * 4);
  __bf16* gated  = (__bf16*)alloc(TV * 2);

  f32_to_bf16_kernel<<<(int)((TE + 255) / 256), 256, 0, stream>>>(hs, hs_bf, TE);

  auto tpose = [&](const float* s, __bf16* d, int rows, int cols) {
    transpose_f32_to_bf16<<<(rows / 32) * (cols / 32), 256, 0, stream>>>(s, d, rows, cols);
  };
  tpose(Wq,   wqt,   E_DIM, E_DIM);
  tpose(Wk,   wkt,   E_DIM, E_DIM);
  tpose(Wv,   wvt,   E_DIM, V_DIM);
  tpose(Wg,   wgt,   E_DIM, V_DIM);
  tpose(Wout, woutt, V_DIM, E_DIM);

  gemm_bf16_kernel<E_DIM, E_DIM><<<(T_LEN / 128) * (E_DIM / 64), 128, 0, stream>>>(hs_bf, wqt, q_f);
  gemm_bf16_kernel<E_DIM, E_DIM><<<(T_LEN / 128) * (E_DIM / 64), 128, 0, stream>>>(hs_bf, wkt, k_f);
  gemm_bf16_kernel<V_DIM, E_DIM><<<(T_LEN / 128) * (V_DIM / 64), 128, 0, stream>>>(hs_bf, wvt, v_f);
  gemm_bf16_kernel<V_DIM, E_DIM><<<(T_LEN / 128) * (V_DIM / 64), 128, 0, stream>>>(hs_bf, wgt, g_f);

  rope_kernel<<<(int)(TE / 2 / 256), 256, 0, stream>>>(q_f, k_f, sn, cs, qr_bf, kr_bf);
  v_transpose_kernel<<<H_NUM * (HD_DIM / 32) * (T_LEN / 32), 256, 0, stream>>>(v_f, vt_bf);

  retention_kernel<<<H_NUM * 32, 128, 0, stream>>>(qr_bf, kr_bf, vt_bf, dm, o_f);

  gate_norm_kernel<<<T_LEN * H_NUM, 64, 0, stream>>>(o_f, g_f, gnw, gated);

  gemm_bf16_kernel<E_DIM, V_DIM><<<(T_LEN / 128) * (E_DIM / 64), 128, 0, stream>>>(gated, woutt, out);
}
